// MultiHeadVer2_3444563771628
// MI455X (gfx1250) — matrix operand order repaired and checked by reading, not compiled, not run on hardware
//
#include <hip/hip_runtime.h>
#include <hip/hip_bf16.h>

// Problem constants (match reference)
#define BB 2
#define TT 4096
#define CC 512
#define HH 8
#define DHD 64
#define MM (BB * TT)   // 8192 rows

typedef __attribute__((ext_vector_type(16))) __bf16 v16bf;
typedef __attribute__((ext_vector_type(8)))  float  v8f;
typedef __attribute__((ext_vector_type(8)))  unsigned short u16x8;
typedef __attribute__((ext_vector_type(4)))  float  f32x4;

union BFrag {
    v16bf v;
    u16x8 h[2];
    unsigned short u[16];
};

__device__ __forceinline__ unsigned short f2bf(float f) {
    unsigned int u = __float_as_uint(f);
    u += 0x7FFFu + ((u >> 16) & 1u);   // round-to-nearest-even
    return (unsigned short)(u >> 16);
}

__device__ __forceinline__ v8f wmma_bf16(const BFrag& a, const BFrag& b, v8f c) {
    // (neg_a, A, neg_b, B, c_mod, C, reuse_a, reuse_b)
    return __builtin_amdgcn_wmma_f32_16x16x32_bf16(
        false, a.v, false, b.v, (short)0, c, false, false);
}

// ---------------------------------------------------------------------------
// Kernel 1: fp32 -> bf16 conversion (vectorized x4)
// ---------------------------------------------------------------------------
__global__ void cvt_f32_bf16(const float* __restrict__ src,
                             unsigned short* __restrict__ dst, int n4) {
    int i = blockIdx.x * blockDim.x + threadIdx.x;
    if (i >= n4) return;
    f32x4 v = *(const f32x4*)(src + 4 * i);
    unsigned long long w = (unsigned long long)f2bf(v.x) |
                           ((unsigned long long)f2bf(v.y) << 16) |
                           ((unsigned long long)f2bf(v.z) << 32) |
                           ((unsigned long long)f2bf(v.w) << 48);
    *(unsigned long long*)(dst + 4 * i) = w;
}

// ---------------------------------------------------------------------------
// Kernel 2: WMMA GEMM  Y = A(M x 512) * W(512 x 512)^T, bf16 in, f32 acc.
// Per wave: 32(M) x 64(N) tile (2 M-subtiles share 4 B fragments), K step 32.
// All fragments of a K-step staged in registers before the 8 WMMAs so the
// unrolled loop can software-pipeline loads under matrix ops.
// mode 0: write Q   -> [B,H,T,DH] bf16
// mode 1: write K   -> [B,H,T,DH] bf16
// mode 2: write V^T -> [B,H,DH,T] bf16
// mode 3: write f32 out[M x 512] = acc + bias[n]
// ---------------------------------------------------------------------------
__global__ __launch_bounds__(256) void gemm_wmma(
    const unsigned short* __restrict__ A,
    const unsigned short* __restrict__ W,
    unsigned short* __restrict__ outb,
    float* __restrict__ outf,
    const float* __restrict__ bias,
    int mode)
{
    const int lane = threadIdx.x & 31;
    const int hl   = lane >> 4;          // half-wave: 0 = lanes 0-15, 1 = 16-31
    const int lm   = lane & 15;
    const int wave = blockIdx.x * (blockDim.x >> 5) + (threadIdx.x >> 5);
    const int nt   = wave & 7;           // 8 N-tiles of 64
    const int mt   = wave >> 3;          // M-tiles of 32
    const int m0   = mt * 32;
    const int n0   = nt * 64;

    v8f acc[2][4] = {};

    // ISA 16-bit A layout: row = m0+lm (+16), K halves at kb and kb+16
    const unsigned short* aptr0 = A + (m0 + lm) * CC + hl * 8;
    const unsigned short* aptr1 = aptr0 + 16 * CC;
    const int koff = hl * 8;             // ISA 16-bit B layout: K halves at kb and kb+16

#pragma unroll
    for (int k0 = 0; k0 < CC; k0 += 32) {
        BFrag a0, a1, bfr[4];
        a0.h[0] = *(const u16x8*)(aptr0 + k0);
        a0.h[1] = *(const u16x8*)(aptr0 + k0 + 16);
        a1.h[0] = *(const u16x8*)(aptr1 + k0);
        a1.h[1] = *(const u16x8*)(aptr1 + k0 + 16);
#pragma unroll
        for (int j = 0; j < 4; ++j) {
            const unsigned short* wp = W + (n0 + j * 16 + lm) * CC + k0 + koff;
            bfr[j].h[0] = *(const u16x8*)(wp);
            bfr[j].h[1] = *(const u16x8*)(wp + 16);
        }
#pragma unroll
        for (int j = 0; j < 4; ++j) {
            acc[0][j] = wmma_bf16(a0, bfr[j], acc[0][j]);
            acc[1][j] = wmma_bf16(a1, bfr[j], acc[1][j]);
        }
    }

    // Writeback.  C/D layout: lane holds col N=lm, rows M = r + 8*hl.
#pragma unroll
    for (int mt2 = 0; mt2 < 2; ++mt2) {
#pragma unroll
        for (int s = 0; s < 4; ++s) {
            const int n = n0 + s * 16 + lm;
            float bv = 0.f;
            if (mode == 3) bv = bias[n];
#pragma unroll
            for (int r = 0; r < 8; ++r) {
                const int m = m0 + mt2 * 16 + r + hl * 8;
                const float val = acc[mt2][s][r];
                if (mode == 3) {
                    outf[m * CC + n] = val + bv;
                } else {
                    const int b = m >> 12;            // m / T
                    const int t = m & (TT - 1);
                    const int h = n >> 6;             // n / DH
                    const int d = n & (DHD - 1);
                    int idx;
                    if (mode == 2) idx = ((b * HH + h) * DHD + d) * TT + t;  // V^T
                    else           idx = ((b * HH + h) * TT + t) * DHD + d;  // Q / K
                    outb[idx] = f2bf(val);
                }
            }
        }
    }
}

// ---------------------------------------------------------------------------
// Kernel 3: causal flash attention, one (b,h) 64-row block per workgroup,
// 16 query rows per wave, 64 keys per chunk, online softmax in registers.
// ---------------------------------------------------------------------------
__global__ __launch_bounds__(128) void flash_attn(
    const unsigned short* __restrict__ qb,   // [B*H, T, DH] bf16
    const unsigned short* __restrict__ kbuf, // [B*H, T, DH] bf16
    const unsigned short* __restrict__ vtb,  // [B*H, DH, T] bf16
    unsigned short* __restrict__ ao)         // [B*T, C] bf16
{
    __shared__ __align__(16) unsigned short pbuf[4][16 * 64];

    const int lane = threadIdx.x & 31;
    const int wv   = threadIdx.x >> 5;       // wave in block (0..3)
    const int hl   = lane >> 4;
    const int lm   = lane & 15;

    const int bh = blockIdx.x / (TT / 64);
    const int tb = blockIdx.x % (TT / 64);
    const int b  = bh / HH;
    const int h  = bh % HH;
    const int i0 = tb * 64 + wv * 16;        // first query row of this wave

    const unsigned short* q  = qb   + bh * (TT * DHD);
    const unsigned short* kk = kbuf + bh * (TT * DHD);
    const unsigned short* vt = vtb  + bh * (DHD * TT);

    // Loop-invariant Q A-fragments (d-chunks 0..31 and 32..63)
    BFrag qa[2];
    {
        const unsigned short* qp = q + (i0 + lm) * DHD + hl * 8;
        qa[0].h[0] = *(const u16x8*)(qp);
        qa[0].h[1] = *(const u16x8*)(qp + 16);
        qa[1].h[0] = *(const u16x8*)(qp + 32);
        qa[1].h[1] = *(const u16x8*)(qp + 48);
    }

    float mrow[8], lrow[8], alpha[8];
    v8f o[4] = {};
#pragma unroll
    for (int r = 0; r < 8; ++r) { mrow[r] = -1e30f; lrow[r] = 0.f; }

    const float scale = 0.044194173824159216f;   // C^-0.5 = 512^-0.5
    const int nchunk = (i0 + 15) / 64 + 1;       // causal trip count (64-key chunks)
    unsigned short* pb = pbuf[wv];

    for (int ch = 0; ch < nchunk; ++ch) {
        const int j0 = ch * 64;

        if (ch + 1 < nchunk) {
            __builtin_prefetch(kk + (j0 + 64 + lane * 2) * DHD, 0, 0);
            __builtin_prefetch(vt + lane * 2 * TT + j0 + 64, 0, 0);
        }

        // ---- S = Q * K^T : 4 column sets of 16 keys, K-dim = DH in 2 chunks.
        //      Batch the 4 B-fragment loads per d-chunk, then 4 WMMAs.
        v8f s4[4] = {};
#pragma unroll
        for (int dc = 0; dc < 2; ++dc) {
            BFrag kf[4];
#pragma unroll
            for (int s = 0; s < 4; ++s) {
                const unsigned short* kp =
                    kk + (j0 + s * 16 + lm) * DHD + dc * 32 + hl * 8;
                kf[s].h[0] = *(const u16x8*)(kp);
                kf[s].h[1] = *(const u16x8*)(kp + 16);
            }
#pragma unroll
            for (int s = 0; s < 4; ++s)
                s4[s] = wmma_bf16(qa[dc], kf[s], s4[s]);
        }

        // ---- scale, causal mask, online softmax; stage P into LDS directly.
        //      C/D layout: row = r + 8*hl, col set s at j0 + s*16 + lm.
#pragma unroll
        for (int r = 0; r < 8; ++r) {
            const int irow = i0 + r + hl * 8;
            const int c0 = j0 + lm;
            float x0 = s4[0][r] * scale;  if (c0 > irow)      x0 = -1e30f;
            float x1 = s4[1][r] * scale;  if (c0 + 16 > irow) x1 = -1e30f;
            float x2 = s4[2][r] * scale;  if (c0 + 32 > irow) x2 = -1e30f;
            float x3 = s4[3][r] * scale;  if (c0 + 48 > irow) x3 = -1e30f;
            float mx = fmaxf(fmaxf(x0, x1), fmaxf(x2, x3));
#pragma unroll
            for (int off = 1; off < 16; off <<= 1)
                mx = fmaxf(mx, __shfl_xor(mx, off, 32));
            const float mnew = fmaxf(mrow[r], mx);
            const float al   = __expf(mrow[r] - mnew);
            const float e0 = __expf(x0 - mnew);
            const float e1 = __expf(x1 - mnew);
            const float e2 = __expf(x2 - mnew);
            const float e3 = __expf(x3 - mnew);
            float rs = (e0 + e1) + (e2 + e3);
#pragma unroll
            for (int off = 1; off < 16; off <<= 1)
                rs += __shfl_xor(rs, off, 32);
            lrow[r]  = lrow[r] * al + rs;
            mrow[r]  = mnew;
            alpha[r] = al;
            const int row = r + hl * 8;
            pb[row * 64 + lm]      = f2bf(e0);
            pb[row * 64 + 16 + lm] = f2bf(e1);
            pb[row * 64 + 32 + lm] = f2bf(e2);
            pb[row * 64 + 48 + lm] = f2bf(e3);
        }
        asm volatile("s_wait_dscnt 0" ::: "memory");  // LDS stores visible before reads

        // ---- reload P as two A-fragments (key chunks 0..31 / 32..63)
        BFrag pf[2];
#pragma unroll
        for (int c = 0; c < 2; ++c) {
            const unsigned short* pr = pb + lm * 64 + c * 32 + hl * 8;
            pf[c].h[0] = *(const u16x8*)(pr);
            pf[c].h[1] = *(const u16x8*)(pr + 16);
        }

        // ---- O = O*alpha + P @ V   (4 output d-sets x 2 key chunks)
#pragma unroll
        for (int s = 0; s < 4; ++s) {
#pragma unroll
            for (int r = 0; r < 8; ++r) o[s][r] *= alpha[r];
        }
#pragma unroll
        for (int s = 0; s < 4; ++s) {
            BFrag vf0, vf1;
            const unsigned short* vp = vt + (s * 16 + lm) * TT + j0 + hl * 8;
            vf0.h[0] = *(const u16x8*)(vp);       vf0.h[1] = *(const u16x8*)(vp + 16);
            vf1.h[0] = *(const u16x8*)(vp + 32);  vf1.h[1] = *(const u16x8*)(vp + 48);
            o[s] = wmma_bf16(pf[0], vf0, o[s]);
            o[s] = wmma_bf16(pf[1], vf1, o[s]);
        }
    }

    // ---- normalize and write attention output as (B,T,C) bf16
#pragma unroll
    for (int r = 0; r < 8; ++r) {
        const float inv = 1.0f / lrow[r];
        const int t = i0 + r + hl * 8;
        const int base = (b * TT + t) * CC + h * DHD;
#pragma unroll
        for (int s = 0; s < 4; ++s) {
            ao[base + s * 16 + lm] = f2bf(o[s][r] * inv);
        }
    }
}

// ---------------------------------------------------------------------------
extern "C" void kernel_launch(void* const* d_in, const int* in_sizes, int n_in,
                              void* d_out, int out_size, void* d_ws, size_t ws_size,
                              hipStream_t stream) {
    const float* x  = (const float*)d_in[0];
    const float* Wq = (const float*)d_in[1];
    const float* Wk = (const float*)d_in[2];
    const float* Wv = (const float*)d_in[3];
    const float* Wp = (const float*)d_in[4];
    const float* bp = (const float*)d_in[5];
    float* out = (float*)d_out;

    char* ws = (char*)d_ws;
    const size_t actBytes = (size_t)MM * CC * sizeof(unsigned short);   // 8 MB
    const size_t wBytes   = (size_t)CC * CC * sizeof(unsigned short);   // 512 KB

    unsigned short* xb  = (unsigned short*)ws;  ws += actBytes;
    unsigned short* wqb = (unsigned short*)ws;  ws += wBytes;
    unsigned short* wkb = (unsigned short*)ws;  ws += wBytes;
    unsigned short* wvb = (unsigned short*)ws;  ws += wBytes;
    unsigned short* wpb = (unsigned short*)ws;  ws += wBytes;
    unsigned short* qd  = (unsigned short*)ws;  ws += actBytes;
    unsigned short* kd  = (unsigned short*)ws;  ws += actBytes;
    unsigned short* vtd = (unsigned short*)ws;  ws += actBytes;
    unsigned short* aod = (unsigned short*)ws;  ws += actBytes;

    // 1) fp32 -> bf16
    {
        const int nx4 = (MM * CC) / 4;
        cvt_f32_bf16<<<(nx4 + 255) / 256, 256, 0, stream>>>(x, xb, nx4);
        const int nw4 = (CC * CC) / 4;
        cvt_f32_bf16<<<(nw4 + 255) / 256, 256, 0, stream>>>(Wq, wqb, nw4);
        cvt_f32_bf16<<<(nw4 + 255) / 256, 256, 0, stream>>>(Wk, wkb, nw4);
        cvt_f32_bf16<<<(nw4 + 255) / 256, 256, 0, stream>>>(Wv, wvb, nw4);
        cvt_f32_bf16<<<(nw4 + 255) / 256, 256, 0, stream>>>(Wp, wpb, nw4);
    }

    // 2) Q/K/V projections (WMMA): (M/32)*(512/64) = 2048 waves, 8 waves/block
    const int gemmBlocks = (MM / 32) * (CC / 64) / 8;   // 256
    gemm_wmma<<<gemmBlocks, 256, 0, stream>>>(xb, wqb, qd,  nullptr, nullptr, 0);
    gemm_wmma<<<gemmBlocks, 256, 0, stream>>>(xb, wkb, kd,  nullptr, nullptr, 1);
    gemm_wmma<<<gemmBlocks, 256, 0, stream>>>(xb, wvb, vtd, nullptr, nullptr, 2);

    // 3) causal flash attention: B*H*(T/64) = 1024 blocks of 4 waves
    flash_attn<<<BB * HH * (TT / 64), 128, 0, stream>>>(qd, kd, vtd, aod);

    // 4) output projection + bias (f32 out)
    gemm_wmma<<<gemmBlocks, 256, 0, stream>>>(aod, wpb, nullptr, out, bp, 3);
}
